// MultiLevelSelfAttentionTransformer_41669772706436
// MI455X (gfx1250) — matrix operand order repaired and checked by reading, not compiled, not run on hardware
//
#include <hip/hip_runtime.h>
#include <cstdint>
#include <cstddef>

// ---------------- CDNA5 WMMA types ----------------
typedef __attribute__((ext_vector_type(16))) __bf16 v16bf;
typedef __attribute__((ext_vector_type(8)))  __bf16 v8bf;
typedef __attribute__((ext_vector_type(8)))  float  v8f;

union V16 { v16bf v; v8bf h[2]; unsigned short u[16]; };

__device__ __forceinline__ unsigned short f2bf(float f) {
    unsigned u = __float_as_uint(f);
    u += 0x7FFFu + ((u >> 16) & 1u);          // round-to-nearest-even
    return (unsigned short)(u >> 16);
}

__device__ __forceinline__ float gelu_exact(float x) {
    return 0.5f * x * (1.0f + erff(x * 0.70710678118654752f));
}

// Async global->LDS copy of 16B (CDNA5). lds = byte offset in workgroup LDS
// (low 32 bits of the generic pointer), ga = 64-bit global address.
__device__ __forceinline__ void async_cp_b128(unsigned lds, unsigned long long ga) {
    asm volatile("global_load_async_to_lds_b128 %0, %1, off"
                 :: "v"(lds), "v"(ga) : "memory");
}

// ---------------- Problem constants ----------------
static constexpr int LNUM = 4;
static constexpr int DMODEL = 1024;
static constexpr int TDIM = 3 * DMODEL;   // 3072
static constexpr int MLP = 4096;
static constexpr int SEQ = 1024;
static constexpr int BATCH = 2;
static constexpr int NH = 16;
static constexpr int HD = 64;
static constexpr int ROWS = BATCH * SEQ;  // 2048
static constexpr float EPSLN = 1e-5f;

// LDS strides (bf16 elements): fragment loads stay 16B aligned
static constexpr int LDK32 = 40;  // 32 + 8 pad  (80 B rows)
static constexpr int LDD64 = 72;  // 64 + 8 pad  (144 B rows)
static constexpr int STAGE = 128 * LDK32;   // elements per tile stage

// =====================================================================
// cvt fp32 -> bf16 (vector of 4 per thread, grid-stride)
// =====================================================================
__global__ __launch_bounds__(256)
void cvt_f32_bf16(const float* __restrict__ in, unsigned short* __restrict__ out, long long n4)
{
    long long i = (long long)blockIdx.x * blockDim.x + threadIdx.x;
    long long stride = (long long)gridDim.x * blockDim.x;
    for (; i < n4; i += stride) {
        const float4 f = *(const float4*)&in[i * 4];
        ushort4 o;
        o.x = f2bf(f.x); o.y = f2bf(f.y); o.z = f2bf(f.z); o.w = f2bf(f.w);
        *(ushort4*)&out[i * 4] = o;
    }
}

// =====================================================================
// GEMM: Y[M,N] = act( Xb[M,K](bf16) @ Wb[N,K](bf16)^T + bias (+ res) )
// BM=BN=128, BK=32, 256 threads = 8 waves (4x2), wave computes 32x64.
// Double-buffered LDS staged with global_load_async_to_lds_b128.
// Steady-state loop is branch-free: issue -> wait(4) -> barrier -> wmma.
// =====================================================================
__global__ __launch_bounds__(256)
void gemm_bf16_wmma(const unsigned short* __restrict__ Xb,
                    const unsigned short* __restrict__ Wb,
                    const float* __restrict__ bias, const float* __restrict__ res,
                    float* __restrict__ Yf, unsigned short* __restrict__ Yb,
                    int M, int N, int K, int act)
{
    (void)M;
    __shared__ unsigned short sA[2 * STAGE];
    __shared__ unsigned short sB[2 * STAGE];

    const int tid = threadIdx.x;
    const int lane = tid & 31;
    const int hf   = lane >> 4;
    const int l16  = lane & 15;
    const int wave = tid >> 5;
    const int wm = wave >> 1;        // 0..3
    const int wn = wave & 1;         // 0..1

    const int m0 = blockIdx.x * 128;
    const int n0 = blockIdx.y * 128;

    const unsigned aBase = (unsigned)(size_t)(void*)sA;
    const unsigned bBase = (unsigned)(size_t)(void*)sB;

    // Per-thread staging slots: 512 16B-chunks per tile, 2 per thread per tile
    const int r0 = (tid + 0)   >> 2, c0 = (tid + 0)   & 3;
    const int r1 = (tid + 256) >> 2, c1 = (tid + 256) & 3;

    v8f acc[2][4];
#pragma unroll
    for (int i = 0; i < 2; ++i)
#pragma unroll
        for (int j = 0; j < 4; ++j)
#pragma unroll
            for (int e = 0; e < 8; ++e) acc[i][j][e] = 0.0f;

    int aoff[2], boff[4];
#pragma unroll
    for (int i = 0; i < 2; ++i) aoff[i] = (wm * 32 + i * 16 + l16) * LDK32 + hf * 8;
#pragma unroll
    for (int j = 0; j < 4; ++j) boff[j] = (wn * 64 + j * 16 + l16) * LDK32 + hf * 8;

    const int nk = K / 32;

    auto issue = [&](int s, int k0) {
        const unsigned sb = (unsigned)(s * STAGE * 2);   // byte offset of stage
        async_cp_b128(aBase + sb + (unsigned)(r0 * 80 + c0 * 16),
                      (unsigned long long)(size_t)&Xb[(size_t)(m0 + r0) * K + k0 + c0 * 8]);
        async_cp_b128(aBase + sb + (unsigned)(r1 * 80 + c1 * 16),
                      (unsigned long long)(size_t)&Xb[(size_t)(m0 + r1) * K + k0 + c1 * 8]);
        async_cp_b128(bBase + sb + (unsigned)(r0 * 80 + c0 * 16),
                      (unsigned long long)(size_t)&Wb[(size_t)(n0 + r0) * K + k0 + c0 * 8]);
        async_cp_b128(bBase + sb + (unsigned)(r1 * 80 + c1 * 16),
                      (unsigned long long)(size_t)&Wb[(size_t)(n0 + r1) * K + k0 + c1 * 8]);
    };

    auto compute = [&](int s) {
        const int so = s * STAGE;
        V16 a[2], b[4];
#pragma unroll
        for (int i = 0; i < 2; ++i) {
            a[i].h[0] = *(const v8bf*)&sA[so + aoff[i]];
            a[i].h[1] = *(const v8bf*)&sA[so + aoff[i] + 16];
        }
#pragma unroll
        for (int j = 0; j < 4; ++j) {
            b[j].h[0] = *(const v8bf*)&sB[so + boff[j]];
            b[j].h[1] = *(const v8bf*)&sB[so + boff[j] + 16];
        }
#pragma unroll
        for (int i = 0; i < 2; ++i)
#pragma unroll
            for (int j = 0; j < 4; ++j)
                acc[i][j] = __builtin_amdgcn_wmma_f32_16x16x32_bf16(
                    false, a[i].v, false, b[j].v, (short)0, acc[i][j], false, false);
    };

    issue(0, 0);
    for (int kk = 0; kk < nk - 1; ++kk) {            // steady state, branch-free
        issue((kk + 1) & 1, (kk + 1) * 32);
        asm volatile("s_wait_asynccnt 0x4" ::: "memory");  // previous stage landed
        __syncthreads();
        compute(kk & 1);
        __syncthreads();
    }
    asm volatile("s_wait_asynccnt 0x0" ::: "memory");
    __syncthreads();
    compute((nk - 1) & 1);

    // Epilogue
#pragma unroll
    for (int i = 0; i < 2; ++i)
#pragma unroll
        for (int j = 0; j < 4; ++j) {
            const int n = n0 + wn * 64 + j * 16 + l16;
            const float bs = bias[n];
#pragma unroll
            for (int e = 0; e < 8; ++e) {
                const int m = m0 + wm * 32 + i * 16 + e + 8 * hf;
                float v = acc[i][j][e] + bs;
                if (res) v += res[(size_t)m * N + n];
                if (act == 1) v = gelu_exact(v);
                if (Yf) Yf[(size_t)m * N + n] = v;
                if (Yb) Yb[(size_t)m * N + n] = f2bf(v);
            }
        }
}

// =====================================================================
// Flash attention (full or banded), bf16 QKV in, bf16 out.
// K tile staged via async-to-LDS; V tile loaded 16B-vectorized and
// transposed into LDS. grid = (S/128, H, B); 8 waves x 16 query rows.
// =====================================================================
__global__ __launch_bounds__(256)
void flash_attn_wmma(const unsigned short* __restrict__ qkv,
                     unsigned short* __restrict__ out, int win)
{
    __shared__ unsigned short sK[32 * LDD64];        // K tile  [t][d]
    __shared__ unsigned short sVt[64 * LDK32];       // V tile transposed [d][t]
    __shared__ unsigned short sP[8 * 16 * LDK32];    // per-wave P staging

    const int tid = threadIdx.x;
    const int lane = tid & 31;
    const int hf = lane >> 4;
    const int l16 = lane & 15;
    const int w = tid >> 5;

    const int m0 = blockIdx.x * 128;
    const int h  = blockIdx.y;
    const int b  = blockIdx.z;

    const unsigned kLds = (unsigned)(size_t)(void*)sK;

    // Q fragments for this wave (rows m0+w*16 .. +15), direct 16B bf16 loads
    const int mq = m0 + w * 16 + l16;
    const unsigned short* qrow = &qkv[((size_t)b * SEQ + mq) * TDIM + h * HD];
    V16 aq[2];
#pragma unroll
    for (int di = 0; di < 2; ++di) {
        aq[di].h[0] = *(const v8bf*)&qrow[di * 32 + hf * 8];
        aq[di].h[1] = *(const v8bf*)&qrow[di * 32 + 16 + hf * 8];
    }

    float run_m[8], run_l[8];
    v8f accO[4];
#pragma unroll
    for (int e = 0; e < 8; ++e) { run_m[e] = -3.0e38f; run_l[e] = 0.0f; }
#pragma unroll
    for (int t = 0; t < 4; ++t)
#pragma unroll
        for (int e = 0; e < 8; ++e) accO[t][e] = 0.0f;

    const unsigned short* kb = &qkv[((size_t)b * SEQ) * TDIM + DMODEL + h * HD];
    const unsigned short* vb = &qkv[((size_t)b * SEQ) * TDIM + 2 * DMODEL + h * HD];
    const int aoffp = (w * 16 + l16) * LDK32 + hf * 8;

    // V staging slot for this thread: row t, 8-elem chunk c8
    const int vt = tid >> 3, vc8 = tid & 7;

    for (int t0 = 0; t0 < SEQ; t0 += 32) {
        if (win >= 0) {   // block-uniform band skip
            if (t0 + 31 < m0 - win || t0 > m0 + 127 + win) continue;
        }
        __syncthreads();
        // K tile: pure strided copy -> async to LDS (128 x 16B chunks)
        if (tid < 128) {
            const int r = tid >> 2, c = tid & 3;
            async_cp_b128(kLds + (unsigned)(r * (LDD64 * 2) + c * 16),
                          (unsigned long long)(size_t)&kb[(size_t)(t0 + r) * TDIM + c * 8]);
        }
        // V tile: vector load 8 bf16, transpose-scatter into LDS
        {
            union { uint4 q; unsigned short s[8]; } u;
            u.q = *(const uint4*)&vb[(size_t)(t0 + vt) * TDIM + vc8 * 8];
#pragma unroll
            for (int i = 0; i < 8; ++i)
                sVt[(vc8 * 8 + i) * LDK32 + vt] = u.s[i];
        }
        asm volatile("s_wait_asynccnt 0x0" ::: "memory");
        __syncthreads();

        // scores: two 16x16 tiles, contraction d=64
        v8f sc[2];
#pragma unroll
        for (int tc = 0; tc < 2; ++tc) {
            const int bofs = (tc * 16 + l16) * LDD64 + hf * 8;
            V16 b0, b1;
            b0.h[0] = *(const v8bf*)&sK[bofs];        b0.h[1] = *(const v8bf*)&sK[bofs + 16];
            b1.h[0] = *(const v8bf*)&sK[bofs + 32];   b1.h[1] = *(const v8bf*)&sK[bofs + 48];
            v8f z;
#pragma unroll
            for (int e = 0; e < 8; ++e) z[e] = 0.0f;
            v8f s = __builtin_amdgcn_wmma_f32_16x16x32_bf16(false, aq[0].v, false, b0.v, (short)0, z, false, false);
            sc[tc] = __builtin_amdgcn_wmma_f32_16x16x32_bf16(false, aq[1].v, false, b1.v, (short)0, s, false, false);
#pragma unroll
            for (int e = 0; e < 8; ++e) sc[tc][e] *= 0.125f;   // 1/sqrt(HD)
        }

        if (win >= 0) {
#pragma unroll
            for (int tc = 0; tc < 2; ++tc) {
                const int tg = t0 + tc * 16 + l16;
#pragma unroll
                for (int e = 0; e < 8; ++e) {
                    const int mg = m0 + w * 16 + e + 8 * hf;
                    int dlt = mg - tg; if (dlt < 0) dlt = -dlt;
                    if (dlt > win) sc[tc][e] = -1.0e30f;
                }
            }
        }

        // online softmax per row (e + 8*hf)
#pragma unroll
        for (int e = 0; e < 8; ++e) {
            float tm = fmaxf(sc[0][e], sc[1][e]);
#pragma unroll
            for (int msk = 8; msk >= 1; msk >>= 1) tm = fmaxf(tm, __shfl_xor(tm, msk, 16));
            const float nm = fmaxf(run_m[e], tm);
            const float fscale = __expf(run_m[e] - nm);
            const float p0 = __expf(sc[0][e] - nm);
            const float p1 = __expf(sc[1][e] - nm);
            float rs = p0 + p1;
#pragma unroll
            for (int msk = 8; msk >= 1; msk >>= 1) rs += __shfl_xor(rs, msk, 16);
            run_l[e] = run_l[e] * fscale + rs;
            run_m[e] = nm;
#pragma unroll
            for (int t = 0; t < 4; ++t) accO[t][e] *= fscale;
            const int r = e + 8 * hf;
            sP[(w * 16 + r) * LDK32 + l16]      = f2bf(p0);
            sP[(w * 16 + r) * LDK32 + 16 + l16] = f2bf(p1);
        }

        // P @ V (contraction 32)
        V16 ap;
        ap.h[0] = *(const v8bf*)&sP[aoffp];
        ap.h[1] = *(const v8bf*)&sP[aoffp + 16];
#pragma unroll
        for (int dt = 0; dt < 4; ++dt) {
            const int bofv = (dt * 16 + l16) * LDK32 + hf * 8;
            V16 bv;
            bv.h[0] = *(const v8bf*)&sVt[bofv];
            bv.h[1] = *(const v8bf*)&sVt[bofv + 16];
            accO[dt] = __builtin_amdgcn_wmma_f32_16x16x32_bf16(false, ap.v, false, bv.v, (short)0, accO[dt], false, false);
        }
    }

#pragma unroll
    for (int dt = 0; dt < 4; ++dt)
#pragma unroll
        for (int e = 0; e < 8; ++e) {
            const int mg = m0 + w * 16 + e + 8 * hf;
            out[((size_t)b * SEQ + mg) * DMODEL + h * HD + dt * 16 + l16] =
                f2bf(accO[dt][e] / run_l[e]);
        }
}

// =====================================================================
// LayerNorm: yf = (x-mu)*rsqrt(var+eps)*g + b ; optional bf16 copy yb.
// =====================================================================
__global__ __launch_bounds__(256)
void layernorm_rows(const float* __restrict__ x, const float* __restrict__ g,
                    const float* __restrict__ bta, float* __restrict__ yf,
                    unsigned short* __restrict__ yb)
{
    __shared__ float r1[256], r2[256];
    const int row = blockIdx.x;
    const int tid = threadIdx.x;
    const float* xr = &x[(size_t)row * DMODEL];

    float s = 0.f, sq = 0.f;
    float v[4];
#pragma unroll
    for (int i = 0; i < 4; ++i) {
        v[i] = xr[tid + i * 256];
        s += v[i]; sq += v[i] * v[i];
    }
    r1[tid] = s; r2[tid] = sq;
    __syncthreads();
    for (int off = 128; off > 0; off >>= 1) {
        if (tid < off) { r1[tid] += r1[tid + off]; r2[tid] += r2[tid + off]; }
        __syncthreads();
    }
    const float mu = r1[0] * (1.0f / DMODEL);
    const float var = r2[0] * (1.0f / DMODEL) - mu * mu;
    const float rinv = rsqrtf(var + EPSLN);
#pragma unroll
    for (int i = 0; i < 4; ++i) {
        const int c = tid + i * 256;
        const float o = (v[i] - mu) * rinv * g[c] + bta[c];
        yf[(size_t)row * DMODEL + c] = o;
        if (yb) yb[(size_t)row * DMODEL + c] = f2bf(o);
    }
}

// =====================================================================
// Host orchestration
// =====================================================================
extern "C" void kernel_launch(void* const* d_in, const int* in_sizes, int n_in,
                              void* d_out, int out_size, void* d_ws, size_t ws_size,
                              hipStream_t stream)
{
    (void)in_sizes; (void)n_in; (void)out_size; (void)ws_size;

    const float* x      = (const float*)d_in[0];
    const float* w_in1  = (const float*)d_in[1];
    const float* b_in1  = (const float*)d_in[2];
    const float* w_out1 = (const float*)d_in[3];
    const float* b_out1 = (const float*)d_in[4];
    const float* ln1_g  = (const float*)d_in[5];
    const float* ln1_b  = (const float*)d_in[6];
    const float* w_in2  = (const float*)d_in[7];
    const float* b_in2  = (const float*)d_in[8];
    const float* w_out2 = (const float*)d_in[9];
    const float* b_out2 = (const float*)d_in[10];
    const float* ln2_g  = (const float*)d_in[11];
    const float* ln2_b  = (const float*)d_in[12];
    const float* ffn_w1 = (const float*)d_in[13];
    const float* ffn_b1 = (const float*)d_in[14];
    const float* ffn_w2 = (const float*)d_in[15];
    const float* ffn_b2 = (const float*)d_in[16];
    const float* lnf_g  = (const float*)d_in[17];
    const float* lnf_b  = (const float*)d_in[18];

    // ---- bf16 workspace carving ----
    unsigned short* ub = (unsigned short*)d_ws;
    auto takeb = [&](size_t n) { unsigned short* p = ub; ub += n; return p; };
    unsigned short* wb_in1  = takeb((size_t)LNUM * TDIM * DMODEL);
    unsigned short* wb_out1 = takeb((size_t)LNUM * DMODEL * DMODEL);
    unsigned short* wb_in2  = takeb((size_t)LNUM * TDIM * DMODEL);
    unsigned short* wb_out2 = takeb((size_t)LNUM * DMODEL * DMODEL);
    unsigned short* wb_f1   = takeb((size_t)LNUM * MLP * DMODEL);
    unsigned short* wb_f2   = takeb((size_t)LNUM * DMODEL * MLP);
    unsigned short* bufQKVb = takeb((size_t)ROWS * TDIM);
    unsigned short* bufA1b  = takeb((size_t)ROWS * DMODEL);
    unsigned short* bufXb   = takeb((size_t)ROWS * DMODEL);
    unsigned short* bufX2b  = takeb((size_t)ROWS * DMODEL);
    unsigned short* bufATTb = takeb((size_t)ROWS * DMODEL);
    unsigned short* bufHb   = takeb((size_t)ROWS * MLP);
    float* f = (float*)((((size_t)ub) + 3) & ~(size_t)3);
    float* bufT   = f; f += (size_t)ROWS * DMODEL;
    float* bufX   = f; f += (size_t)ROWS * DMODEL;
    float* bufX2  = f; f += (size_t)ROWS * DMODEL;
    float* bufATT = f; f += (size_t)ROWS * DMODEL;

    const dim3 blk(256);

    // ---- one-time conversions: weights + initial x ----
    auto cvt = [&](const float* src, unsigned short* dst, size_t n) {
        long long n4 = (long long)(n / 4);
        int grid = (int)((n4 + 2047) / 2048);
        cvt_f32_bf16<<<grid, blk, 0, stream>>>(src, dst, n4);
    };
    cvt(w_in1,  wb_in1,  (size_t)LNUM * TDIM * DMODEL);
    cvt(w_out1, wb_out1, (size_t)LNUM * DMODEL * DMODEL);
    cvt(w_in2,  wb_in2,  (size_t)LNUM * TDIM * DMODEL);
    cvt(w_out2, wb_out2, (size_t)LNUM * DMODEL * DMODEL);
    cvt(ffn_w1, wb_f1,   (size_t)LNUM * MLP * DMODEL);
    cvt(ffn_w2, wb_f2,   (size_t)LNUM * DMODEL * MLP);
    cvt(x,      bufXb,   (size_t)ROWS * DMODEL);
    hipMemcpyAsync(bufX, x, (size_t)ROWS * DMODEL * sizeof(float),
                   hipMemcpyDeviceToDevice, stream);

    const dim3 gQKV(ROWS / 128, TDIM / 128);
    const dim3 gOUT(ROWS / 128, DMODEL / 128);
    const dim3 gF1 (ROWS / 128, MLP / 128);
    const dim3 gAtt(SEQ / 128, NH, BATCH);

    for (int l = 0; l < LNUM; ++l) {
        const unsigned short* wi1 = wb_in1  + (size_t)l * TDIM * DMODEL;
        const unsigned short* wo1 = wb_out1 + (size_t)l * DMODEL * DMODEL;
        const unsigned short* wi2 = wb_in2  + (size_t)l * TDIM * DMODEL;
        const unsigned short* wo2 = wb_out2 + (size_t)l * DMODEL * DMODEL;
        const unsigned short* fw1 = wb_f1   + (size_t)l * MLP * DMODEL;
        const unsigned short* fw2 = wb_f2   + (size_t)l * DMODEL * MLP;
        const float* bi1 = b_in1  + (size_t)l * TDIM;
        const float* bo1 = b_out1 + (size_t)l * DMODEL;
        const float* bi2 = b_in2  + (size_t)l * TDIM;
        const float* bo2 = b_out2 + (size_t)l * DMODEL;
        const float* fb1 = ffn_b1 + (size_t)l * MLP;
        const float* fb2 = ffn_b2 + (size_t)l * DMODEL;

        // intra-frame full MHA
        gemm_bf16_wmma<<<gQKV, blk, 0, stream>>>(bufXb, wi1, bi1, nullptr, nullptr, bufQKVb, ROWS, TDIM, DMODEL, 0);
        flash_attn_wmma<<<gAtt, blk, 0, stream>>>(bufQKVb, bufA1b, -1);
        gemm_bf16_wmma<<<gOUT, blk, 0, stream>>>(bufA1b, wo1, bo1, bufX, bufT, nullptr, ROWS, DMODEL, DMODEL, 0);
        layernorm_rows<<<ROWS, blk, 0, stream>>>(bufT, ln1_g + (size_t)l * DMODEL, ln1_b + (size_t)l * DMODEL, bufX2, bufX2b);

        // inter-frame banded MHA
        gemm_bf16_wmma<<<gQKV, blk, 0, stream>>>(bufX2b, wi2, bi2, nullptr, nullptr, bufQKVb, ROWS, TDIM, DMODEL, 0);
        flash_attn_wmma<<<gAtt, blk, 0, stream>>>(bufQKVb, bufA1b, 2);
        gemm_bf16_wmma<<<gOUT, blk, 0, stream>>>(bufA1b, wo2, bo2, bufX2, bufT, nullptr, ROWS, DMODEL, DMODEL, 0);
        layernorm_rows<<<ROWS, blk, 0, stream>>>(bufT, ln2_g + (size_t)l * DMODEL, ln2_b + (size_t)l * DMODEL, bufATT, bufATTb);

        // FFN (GELU exact) + residual from attended
        gemm_bf16_wmma<<<gF1, blk, 0, stream>>>(bufATTb, fw1, fb1, nullptr, nullptr, bufHb, ROWS, MLP, DMODEL, 1);
        gemm_bf16_wmma<<<gOUT, blk, 0, stream>>>(bufHb, fw2, fb2, bufATT, bufX, bufXb, ROWS, DMODEL, MLP, 0);
    }

    layernorm_rows<<<ROWS, blk, 0, stream>>>(bufX, lnf_g, lnf_b, (float*)d_out, nullptr);
}
